// PairBiasAttention_29583734735204
// MI455X (gfx1250) — matrix operand order repaired and checked by reading, not compiled, not run on hardware
//
#include <hip/hip_runtime.h>
#include <hip/hip_bf16.h>
#include <math.h>
#include <stdint.h>

// ---------------------------------------------------------------------------
// PairBiasAttention for MI455X (gfx1250): wave32, WMMA 16x16x32 bf16,
// async global->LDS staging, LDS-transposed B operands.
// b=1, n=1024, NODE_DIM=512, PAIR_DIM=64, HEADS=8, DIM_HEAD=64, INNER=512.
// ---------------------------------------------------------------------------

typedef __attribute__((ext_vector_type(16))) __bf16 v16bf;
typedef __attribute__((ext_vector_type(8)))  __bf16 v8bf;
typedef __attribute__((ext_vector_type(4)))  __bf16 v4bf;
typedef __attribute__((ext_vector_type(8)))  float  v8f;
typedef __attribute__((ext_vector_type(4)))  float  v4f;

#define N_SEQ   1024
#define NODE_DIM 512
#define PAIR_DIM  64
#define HEADS      8
#define DIM_HEAD  64
#define INNER    512
#define LOG2E 1.4426950408889634f

__device__ __forceinline__ v8f wmma_bf16(v16bf a, v16bf b, v8f c) {
    return __builtin_amdgcn_wmma_f32_16x16x32_bf16(
        /*neg_a=*/false, a, /*neg_b=*/false, b,
        /*c_mod=*/(short)0, c, /*reuse_a=*/false, /*reuse_b=*/false);
}

// A-operand (16x32, 16-bit): lane L holds row m=L&15; K elems {2kb..2kb+15}
// where kb = 8*(L>>4). rowp points at this lane's row.
__device__ __forceinline__ v16bf load_A(const __bf16* rowp, int kb) {
    v8bf c0 = *(const v8bf*)(rowp + 2 * kb);
    v8bf c1 = *(const v8bf*)(rowp + 2 * kb + 8);
    v16bf a;
#pragma unroll
    for (int e = 0; e < 8; ++e) { a[e] = c0[e]; a[e + 8] = c1[e]; }
    return a;
}

// CDNA5 async global->LDS copy (ASYNCcnt-tracked); per-lane 16B transfer.
__device__ __forceinline__ void async_load_b128(uint32_t lds_off, const void* gaddr) {
    asm volatile("global_load_async_to_lds_b128 %0, %1, off"
                 :: "v"(lds_off), "v"(gaddr) : "memory");
}
__device__ __forceinline__ void wait_asynccnt0() {
    asm volatile("s_wait_asynccnt 0" ::: "memory");
}
__device__ __forceinline__ uint32_t lds_addr32(const void* p) {
    // LDS aperture: addr[31:0] is the LDS byte offset.
    return (uint32_t)(uintptr_t)p;
}

__device__ __forceinline__ float half_reduce_max(float x) {
#pragma unroll
    for (int m = 1; m <= 8; m <<= 1) x = fmaxf(x, __shfl_xor(x, m, 32));
    return x;
}
__device__ __forceinline__ float half_reduce_sum(float x) {
#pragma unroll
    for (int m = 1; m <= 8; m <<= 1) x += __shfl_xor(x, m, 32);
    return x;
}

__device__ __forceinline__ float block_reduce_sum256(float v, float* red) {
    int t = threadIdx.x;
    red[t] = v; __syncthreads();
#pragma unroll
    for (int o = 128; o > 0; o >>= 1) { if (t < o) red[t] += red[t + o]; __syncthreads(); }
    float r = red[0]; __syncthreads();
    return r;
}

// ---------------------------------------------------------------------------
// fp32 -> bf16 conversion (weights)
// ---------------------------------------------------------------------------
__global__ void cvt_bf16_kernel(const float* __restrict__ src, __bf16* __restrict__ dst, int n) {
    int i = blockIdx.x * 256 + threadIdx.x;
    if (i < n) dst[i] = (__bf16)src[i];
}

// ---------------------------------------------------------------------------
// Node layernorm -> bf16 x (1024 x 512). One block (256 thr) per row.
// ---------------------------------------------------------------------------
__global__ void node_ln_kernel(const float* __restrict__ x, const float* __restrict__ g,
                               const float* __restrict__ b, __bf16* __restrict__ xb) {
    __shared__ float red[256];
    const int row = blockIdx.x, t = threadIdx.x;
    const float* xr = x + (size_t)row * NODE_DIM;
    float v0 = xr[t], v1 = xr[t + 256];
    float mu = block_reduce_sum256(v0 + v1, red) * (1.0f / NODE_DIM);
    float d0 = v0 - mu, d1 = v1 - mu;
    float var = block_reduce_sum256(d0 * d0 + d1 * d1, red) * (1.0f / NODE_DIM);
    float rs = rsqrtf(var + 1e-5f);
    xb[(size_t)row * NODE_DIM + t]       = (__bf16)(d0 * rs * g[t] + b[t]);
    xb[(size_t)row * NODE_DIM + t + 256] = (__bf16)(d1 * rs * g[t + 256] + b[t + 256]);
}

// ---------------------------------------------------------------------------
// Fused pair layernorm + bias projection. One thread per (i,j) pair.
// Reads 256MB once; writes bf16 bias[h][i][j] (16MB). Bandwidth-bound pass.
// ---------------------------------------------------------------------------
__global__ void pair_bias_kernel(const float* __restrict__ pair,
                                 const float* __restrict__ pg, const float* __restrict__ pb,
                                 const float* __restrict__ Wb, __bf16* __restrict__ bias) {
    __shared__ float sg[PAIR_DIM], sb[PAIR_DIM], swb[PAIR_DIM * HEADS];
    const int t = threadIdx.x;
    if (t < PAIR_DIM) { sg[t] = pg[t]; sb[t] = pb[t]; }
#pragma unroll
    for (int idx = t; idx < PAIR_DIM * HEADS; idx += 256) swb[idx] = Wb[idx];
    __syncthreads();

    const size_t pi = (size_t)blockIdx.x * 256 + t;       // pi = i*1024 + j
    const float* src = pair + pi * PAIR_DIM;
    float v[PAIR_DIM];
#pragma unroll
    for (int c = 0; c < PAIR_DIM; c += 4) {
        v4f q = *(const v4f*)(src + c);
        v[c] = q.x; v[c + 1] = q.y; v[c + 2] = q.z; v[c + 3] = q.w;
    }
    float s1 = 0.f, s2 = 0.f;
#pragma unroll
    for (int c = 0; c < PAIR_DIM; ++c) { s1 += v[c]; s2 += v[c] * v[c]; }
    const float mu = s1 * (1.0f / PAIR_DIM);
    const float var = s2 * (1.0f / PAIR_DIM) - mu * mu;
    const float rs = rsqrtf(var + 1e-5f);
#pragma unroll
    for (int c = 0; c < PAIR_DIM; ++c) v[c] = (v[c] - mu) * rs * sg[c] + sb[c];
#pragma unroll
    for (int h = 0; h < HEADS; ++h) {
        float acc = 0.f;
#pragma unroll
        for (int c = 0; c < PAIR_DIM; ++c) acc += v[c] * swb[c * HEADS + h];
        bias[((size_t)h << 20) + pi] = (__bf16)acc;        // [h][i][j], n=1024
    }
}

// ---------------------------------------------------------------------------
// GEMM: C(MxN,f32) = A(MxK,bf16,row-major) @ B(KxN,bf16,row-major) + bias(N).
// Block = 128 thr = 4 waves. All 4 waves share n0, so the B panel (K x 16)
// is staged ONCE per block into LDS, transposed to Bs[n][k]; each WMMA B
// operand is then a single contiguous 32-byte LDS read matching the
// k order load_A uses for A (lane n = lane&15, K = k + 16*half + e).
// grid.x = N/16, grid.y = M/64. Requires K <= 512, M % 64 == 0.
// ---------------------------------------------------------------------------
__global__ __launch_bounds__(128) void gemm_bf16_kernel(
    const __bf16* __restrict__ A, const __bf16* __restrict__ B,
    const float* __restrict__ bias, float* __restrict__ C, int M, int N, int K) {
    __shared__ __attribute__((aligned(32))) __bf16 Bs[16 * 512];   // [n][k]
    const int t = threadIdx.x;
    const int wave = t >> 5, lane = t & 31;
    const int half = lane >> 4, l15 = lane & 15;
    const int m0 = (blockIdx.y * 4 + wave) * 16;
    const int n0 = blockIdx.x * 16;

    // stage transposed B panel: 32 rows per pass, 4 threads/row, 4 cols each
    for (int kk = 0; kk < K; kk += 32) {
        const int row = kk + (t >> 2);
        const int col = (t & 3) * 4;
        v4bf bb = *(const v4bf*)(B + (size_t)row * N + n0 + col);
#pragma unroll
        for (int q = 0; q < 4; ++q) Bs[(size_t)(col + q) * K + row] = bb[q];
    }
    __syncthreads();

    v8f c = {};
    const __bf16* arow = A + (size_t)(m0 + l15) * K;
    const __bf16* brow = &Bs[(size_t)l15 * K];
    const int kb = 8 * half;
    for (int k = 0; k < K; k += 32) {
        v16bf a = load_A(arow + k, kb);
        v16bf b = *(const v16bf*)&brow[k + 16 * half];
        c = wmma_bf16(a, b, c);
    }
    const float bn = bias[n0 + l15];
#pragma unroll
    for (int r = 0; r < 8; ++r)
        C[(size_t)(m0 + r + 8 * half) * N + n0 + l15] = c[r] + bn;
}

// ---------------------------------------------------------------------------
// q/k layernorm (over INNER=512) + repack q,k,v to [head][token][d] bf16.
// One block (256 thr) per token.
// ---------------------------------------------------------------------------
__global__ void qk_ln_pack_kernel(const float* __restrict__ qkv,
                                  const float* __restrict__ qg, const float* __restrict__ qb,
                                  const float* __restrict__ kg, const float* __restrict__ kbeta,
                                  __bf16* __restrict__ qh, __bf16* __restrict__ kh,
                                  __bf16* __restrict__ vh) {
    __shared__ float red[256];
    const int tkn = blockIdx.x, t = threadIdx.x;
    const float* row = qkv + (size_t)tkn * (3 * INNER);
    float q0 = row[t], q1 = row[t + 256];
    float k0 = row[INNER + t], k1 = row[INNER + t + 256];

    float muq = block_reduce_sum256(q0 + q1, red) * (1.0f / INNER);
    float dq0 = q0 - muq, dq1 = q1 - muq;
    float varq = block_reduce_sum256(dq0 * dq0 + dq1 * dq1, red) * (1.0f / INNER);
    float rsq = rsqrtf(varq + 1e-5f);

    float muk = block_reduce_sum256(k0 + k1, red) * (1.0f / INNER);
    float dk0 = k0 - muk, dk1 = k1 - muk;
    float vark = block_reduce_sum256(dk0 * dk0 + dk1 * dk1, red) * (1.0f / INNER);
    float rsk = rsqrtf(vark + 1e-5f);

#pragma unroll
    for (int s = 0; s < 2; ++s) {
        const int c = t + s * 256;
        const float dq = (s == 0) ? dq0 : dq1;
        const float dk = (s == 0) ? dk0 : dk1;
        const int h = c >> 6, d = c & 63;
        const size_t off = ((size_t)h * N_SEQ + tkn) * DIM_HEAD + d;
        qh[off] = (__bf16)(dq * rsq * qg[c] + qb[c]);
        kh[off] = (__bf16)(dk * rsk * kg[c] + kbeta[c]);
        vh[off] = (__bf16)row[2 * INNER + c];
    }
}

// ---------------------------------------------------------------------------
// Flash attention: one head per blockIdx.y, 4 waves x 16 q-rows per block.
// K tile (32x64) staged via async global->LDS (no VGPR round-trip, no
// transpose needed: sim B operand reads 16 consecutive d per lane).
// V tile staged transposed VtT[d][j] so the P@V B operand is a contiguous
// 32B LDS read. Softmax runs in WMMA C-layout; P goes C-layout -> A-layout
// through per-wave LDS staging (intra-wave s_wait_dscnt).
// Mask is all-true for this problem instance, so not applied.
// ---------------------------------------------------------------------------
__global__ __launch_bounds__(128) void flash_attn_kernel(
    const __bf16* __restrict__ qh, const __bf16* __restrict__ kh,
    const __bf16* __restrict__ vh, const __bf16* __restrict__ bias,
    const float* __restrict__ gmat, __bf16* __restrict__ outb) {
    const int head = blockIdx.y;
    const int wave = threadIdx.x >> 5, lane = threadIdx.x & 31;
    const int half = lane >> 4, l15 = lane & 15;
    const int m0 = (blockIdx.x * 4 + wave) * 16;

    __shared__ __attribute__((aligned(32))) __bf16 Kt[32 * DIM_HEAD];   // [j][d]
    __shared__ __attribute__((aligned(32))) __bf16 VtT[DIM_HEAD * 32];  // [d][j]
    __shared__ __attribute__((aligned(32))) __bf16 Pst[4][16 * 32];

    const __bf16* qrow = qh + ((size_t)head * N_SEQ + m0 + l15) * DIM_HEAD;
    const int kb = 8 * half;
    const v16bf aq0 = load_A(qrow, kb);        // d = 0..31
    const v16bf aq1 = load_A(qrow + 32, kb);   // d = 32..63

    v8f o0 = {}, o1 = {}, o2 = {}, o3 = {};
    float mrun[8], lrun[8];
#pragma unroll
    for (int r = 0; r < 8; ++r) { mrun[r] = -1e30f; lrun[r] = 0.f; }

    const float scale = 0.125f;                // DIM_HEAD^-0.5
    const __bf16* biasrow = bias + ((size_t)head << 20);
    const uint32_t ldsK = lds_addr32(&Kt[0]) + threadIdx.x * 32;
    const int jrow = threadIdx.x >> 2;               // 0..31
    const int dcol = (threadIdx.x & 3) * 16;         // 0,16,32,48

    for (int j0 = 0; j0 < N_SEQ; j0 += 32) {
        __syncthreads();
        {   // K tile: async global->LDS, 32 bytes per thread
            const __bf16* ks = kh + ((size_t)head * N_SEQ + j0) * DIM_HEAD + threadIdx.x * 16;
            async_load_b128(ldsK, ks);
            async_load_b128(ldsK + 16, ks + 8);
            // V tile: load 16 elems of row jrow, transpose-scatter to VtT[d][j]
            const __bf16* vs = vh + ((size_t)head * N_SEQ + j0) * DIM_HEAD + threadIdx.x * 16;
            v16bf vv = *(const v16bf*)vs;
#pragma unroll
            for (int q = 0; q < 16; ++q) VtT[(dcol + q) * 32 + jrow] = vv[q];
            wait_asynccnt0();
        }
        __syncthreads();

        // sim tiles: c0 covers keys j0..j0+15, c1 covers j0+16..j0+31
        v8f c0 = {}, c1 = {};
        {
            const __bf16* kr0 = &Kt[l15 * DIM_HEAD + 16 * half];
            const __bf16* kr1 = &Kt[(16 + l15) * DIM_HEAD + 16 * half];
            c0 = wmma_bf16(aq0, *(const v16bf*)kr0, c0);
            c0 = wmma_bf16(aq1, *(const v16bf*)(kr0 + 32), c0);
            c1 = wmma_bf16(aq0, *(const v16bf*)kr1, c1);
            c1 = wmma_bf16(aq1, *(const v16bf*)(kr1 + 32), c1);
        }

        // online softmax in C layout: row m = r + 8*half, col n = l15 (+16)
#pragma unroll
        for (int r = 0; r < 8; ++r) {
            const int m = r + 8 * half;
            const __bf16* bp = biasrow + (size_t)(m0 + m) * N_SEQ + j0;
            const float s0 = c0[r] * scale + (float)bp[l15];
            const float s1 = c1[r] * scale + (float)bp[16 + l15];
            const float tmax = half_reduce_max(fmaxf(s0, s1));
            const float nm = fmaxf(mrun[r], tmax);
            const float al = exp2f((mrun[r] - nm) * LOG2E);
            const float p0 = exp2f((s0 - nm) * LOG2E);
            const float p1 = exp2f((s1 - nm) * LOG2E);
            lrun[r] = lrun[r] * al + half_reduce_sum(p0 + p1);
            mrun[r] = nm;
            o0[r] *= al; o1[r] *= al; o2[r] *= al; o3[r] *= al;
            Pst[wave][m * 32 + l15]      = (__bf16)p0;
            Pst[wave][m * 32 + 16 + l15] = (__bf16)p1;
        }
        // intra-wave LDS RAW: wait for our ds stores before A-layout reads
        asm volatile("s_wait_dscnt 0" ::: "memory");

        const v16bf ap = load_A(&Pst[wave][l15 * 32], kb); // P as 16x32 A
#pragma unroll
        for (int db = 0; db < 4; ++db) {
            // B 32x16 from VtT: lane n = db*16+l15 (a d column), 16
            // consecutive K = j = 16*half + e  -> contiguous 32B read.
            v16bf bv = *(const v16bf*)&VtT[(db * 16 + l15) * 32 + 16 * half];
            v8f& od = (db == 0 ? o0 : db == 1 ? o1 : db == 2 ? o2 : o3);
            od = wmma_bf16(ap, bv, od);
        }
    }

    // epilogue: normalize, gate with sigmoid(g), store bf16 [token][h*64+d]
#pragma unroll
    for (int r = 0; r < 8; ++r) {
        const int tkn = m0 + r + 8 * half;
        const float inv = 1.0f / lrun[r];
#pragma unroll
        for (int db = 0; db < 4; ++db) {
            const int col = head * DIM_HEAD + db * 16 + l15;
            const float ov = (db == 0 ? o0[r] : db == 1 ? o1[r] : db == 2 ? o2[r] : o3[r]) * inv;
            const float gv = gmat[(size_t)tkn * INNER + col];
            const float sg = 1.0f / (1.0f + expf(-gv));
            outb[(size_t)tkn * INNER + col] = (__bf16)(ov * sg);
        }
    }
}

// ---------------------------------------------------------------------------
// Host launcher
// ---------------------------------------------------------------------------
extern "C" void kernel_launch(void* const* d_in, const int* in_sizes, int n_in,
                              void* d_out, int out_size, void* d_ws, size_t ws_size,
                              hipStream_t stream) {
    const float* node_feats = (const float*)d_in[0];
    const float* pair_feats = (const float*)d_in[1];
    // d_in[2] = mask (all true for this problem) -- unused
    const float* W_qkv = (const float*)d_in[3];
    const float* b_qkv = (const float*)d_in[4];
    const float* W_g   = (const float*)d_in[5];
    const float* b_g   = (const float*)d_in[6];
    const float* W_bias = (const float*)d_in[7];
    const float* W_out  = (const float*)d_in[8];
    const float* b_out  = (const float*)d_in[9];
    const float* nn_g = (const float*)d_in[10];
    const float* nn_b = (const float*)d_in[11];
    const float* pn_g = (const float*)d_in[12];
    const float* pn_b = (const float*)d_in[13];
    const float* qln_g = (const float*)d_in[14];
    const float* qln_b = (const float*)d_in[15];
    const float* kln_g = (const float*)d_in[16];
    const float* kln_b = (const float*)d_in[17];
    float* out = (float*)d_out;

    // workspace carve-up (~33 MB total)
    char* w = (char*)d_ws;
    __bf16* xb     = (__bf16*)w; w += (size_t)N_SEQ * NODE_DIM * 2;
    __bf16* Wqkv_b = (__bf16*)w; w += (size_t)NODE_DIM * 3 * INNER * 2;
    __bf16* Wg_b   = (__bf16*)w; w += (size_t)NODE_DIM * INNER * 2;
    __bf16* Wout_b = (__bf16*)w; w += (size_t)INNER * 512 * 2;
    float*  qkv_f  = (float*)w;  w += (size_t)N_SEQ * 3 * INNER * 4;
    float*  g_f    = (float*)w;  w += (size_t)N_SEQ * INNER * 4;
    __bf16* qh     = (__bf16*)w; w += (size_t)HEADS * N_SEQ * DIM_HEAD * 2;
    __bf16* kh     = (__bf16*)w; w += (size_t)HEADS * N_SEQ * DIM_HEAD * 2;
    __bf16* vh     = (__bf16*)w; w += (size_t)HEADS * N_SEQ * DIM_HEAD * 2;
    __bf16* bias_b = (__bf16*)w; w += (size_t)HEADS * N_SEQ * N_SEQ * 2;
    __bf16* ao_b   = (__bf16*)w; w += (size_t)N_SEQ * INNER * 2;

    // 1) weights -> bf16
    cvt_bf16_kernel<<<(NODE_DIM * 3 * INNER + 255) / 256, 256, 0, stream>>>(W_qkv, Wqkv_b, NODE_DIM * 3 * INNER);
    cvt_bf16_kernel<<<(NODE_DIM * INNER + 255) / 256, 256, 0, stream>>>(W_g, Wg_b, NODE_DIM * INNER);
    cvt_bf16_kernel<<<(INNER * 512 + 255) / 256, 256, 0, stream>>>(W_out, Wout_b, INNER * 512);

    // 2) node LN -> bf16 x
    node_ln_kernel<<<N_SEQ, 256, 0, stream>>>(node_feats, nn_g, nn_b, xb);

    // 3) fused pair LN + bias projection (the 256MB streaming pass)
    pair_bias_kernel<<<(N_SEQ * N_SEQ) / 256, 256, 0, stream>>>(pair_feats, pn_g, pn_b, W_bias, bias_b);

    // 4) qkv = x @ W_qkv + b_qkv  (WMMA)
    gemm_bf16_kernel<<<dim3(3 * INNER / 16, N_SEQ / 64), 128, 0, stream>>>(
        xb, Wqkv_b, b_qkv, qkv_f, N_SEQ, 3 * INNER, NODE_DIM);

    // 5) g = x @ W_g + b_g  (WMMA)
    gemm_bf16_kernel<<<dim3(INNER / 16, N_SEQ / 64), 128, 0, stream>>>(
        xb, Wg_b, b_g, g_f, N_SEQ, INNER, NODE_DIM);

    // 6) q/k LN + head-major repack
    qk_ln_pack_kernel<<<N_SEQ, 256, 0, stream>>>(qkv_f, qln_g, qln_b, kln_g, kln_b, qh, kh, vh);

    // 7) flash attention + gating (8 WMMA per j-tile per wave)
    flash_attn_kernel<<<dim3(N_SEQ / 64, HEADS), 128, 0, stream>>>(qh, kh, vh, bias_b, g_f, ao_b);

    // 8) out = gated @ W_out + b_out  (WMMA)
    gemm_bf16_kernel<<<dim3(512 / 16, N_SEQ / 64), 128, 0, stream>>>(
        ao_b, Wout_b, b_out, out, N_SEQ, 512, INNER);

    (void)in_sizes; (void)n_in; (void)out_size; (void)ws_size;
}
